// SPVCNN_584115552812
// MI455X (gfx1250) — matrix operand order repaired and checked by reading, not compiled, not run on hardware
//
// SPVCNN forward for MI455X (gfx1250, wave32, WMMA).
// Gather-GEMM sparse conv on v_wmma_f32_16x16x32_bf16 with double-buffered
// async LDS staging (global_load_async_to_lds_b128 / s_wait_asynccnt).
// Per block: 32 rows x 128 cols; per wave: 2x2 subtiles = 4 WMMA per A chunk.
#include <hip/hip_runtime.h>
#include <stdint.h>
#include <stddef.h>

typedef __attribute__((ext_vector_type(16))) __bf16 v16bf;
typedef __attribute__((ext_vector_type(8)))  float  v8f;
typedef __attribute__((ext_vector_type(4)))  int    v4i;

#define EPSBN 1e-5f

// ---------------------------------------------------------------------------
// gfx1250 async copy helpers (ISA cdna5 §15.18 opcode 98)
// ---------------------------------------------------------------------------
__device__ __forceinline__ void async_g2l_b128(void* lds_dst, const void* gsrc) {
  asm volatile("global_load_async_to_lds_b128 %0, %1, off"
               :: "v"((unsigned)(size_t)lds_dst), "v"(gsrc)
               : "memory");
}
__device__ __forceinline__ void wait_async0() {
  asm volatile("s_wait_asynccnt 0" ::: "memory");
}

__device__ __forceinline__ v8f bfwmma(v16bf a, v16bf b, v8f c) {
  return __builtin_amdgcn_wmma_f32_16x16x32_bf16(false, a, false, b, (short)0, c,
                                                 false, false);
}

// ---------------------------------------------------------------------------
// Gather-conv:  out[r,:] = sum_k feat[nbr[k][r],:] @ W[k]
// Epilogue: (+bias) -> BN -> (+residual) -> ReLU, dual f32/bf16 store.
//
// Wt layout (device packed): Wt[((k*nch + cc)*coutp + n)*32 + j] = W[k][cc*32+j][n]
// WMMA bf16 16x16x32 fragment maps (wave32):
//   A: lane L: m=L&15, hi=L>>4, half j -> K = (j>>3)*16 + hi*8 + (j&7)
//   B: lane L: n=L&15, hi=L>>4, half j -> K = (j>>3)*16 + hi*8 + (j&7)
//   C: lane L: n=L&15, vgpr r  -> m = r + 8*hi
// ---------------------------------------------------------------------------
__global__ __launch_bounds__(128) void spconv_wmma(
    const __bf16* __restrict__ feat, int in_stride,
    const __bf16* __restrict__ Wt,
    const int* __restrict__ nbr, int pitch,
    int K, int nch, int n_out, int cout, int coutp,
    float* __restrict__ outF, __bf16* __restrict__ outB, int out_stride, int col_off,
    const float* __restrict__ bias,
    const float* __restrict__ gp, const float* __restrict__ bp,
    const float* __restrict__ mp, const float* __restrict__ vp,
    const float* __restrict__ resid, int rstride, int relu)
{
  __shared__ __align__(16) __bf16 ldsA[2][32 * 32];   // double-buffered 32x32 bf16

  const int tid  = threadIdx.x;
  const int lane = tid & 31;
  const int wave = tid >> 5;
  const int row0 = blockIdx.x * 32;
  const int n0   = blockIdx.y * 128 + wave * 32;      // wave handles cols [n0, n0+32)
  const bool act0 = (n0 < coutp);
  const bool act1 = (n0 + 16 < coutp);

  const int m  = lane & 15;
  const int hi = lane >> 4;

  // staging role: 128 threads x 16B = full 32x32 bf16 tile
  const int sr   = tid >> 2;       // row 0..31
  const int sseg = tid & 3;        // 16B segment
  const int srow = row0 + sr;

  const int T = K * nch;
  int curk = -1, sidx = -1;

  v8f a00 = {}, a01 = {}, a10 = {}, a11 = {};

  auto stage = [&](int t, int b) {
    int k = t / nch, cc = t - k * nch;
    if (k != curk) {
      curk = k;
      sidx = (srow < n_out) ? nbr[(size_t)k * pitch + srow] : -1;
    }
    __bf16* dst = &ldsA[b][sr * 32 + sseg * 8];
    if (sidx >= 0) {
      async_g2l_b128(dst, feat + (size_t)sidx * in_stride + cc * 32 + sseg * 8);
    } else {
      v4i z = {};
      *(v4i*)dst = z;                                  // ds_store_b128 zero fill
    }
  };

  stage(0, 0);
  int buf = 0;
  for (int t = 0; t < T; ++t) {
    wait_async0();          // own wave's async loads into `buf` complete
    __syncthreads();        // everyone's staging of `buf` complete
    if (t + 1 < T) stage(t + 1, buf ^ 1);
    if (act0) {
      union Frag { v16bf v; v4i q[2]; } A0, A1, B0, B1;
      const __bf16* base = &ldsA[buf][0];
      A0.q[0] = *(const v4i*)&base[m * 32 + hi * 8];
      A0.q[1] = *(const v4i*)&base[m * 32 + 16 + hi * 8];
      A1.q[0] = *(const v4i*)&base[(m + 16) * 32 + hi * 8];
      A1.q[1] = *(const v4i*)&base[(m + 16) * 32 + 16 + hi * 8];
      size_t wrow = (size_t)t * coutp;                 // t == k*nch+cc
      size_t wb0 = (wrow + n0 + m) * 32 + hi * 8;
      B0.q[0] = *(const v4i*)&Wt[wb0];
      B0.q[1] = *(const v4i*)&Wt[wb0 + 16];
      __builtin_prefetch(&Wt[wb0 + (size_t)coutp * 32], 0, 1); // global_prefetch_b8
      a00 = bfwmma(A0.v, B0.v, a00);
      a10 = bfwmma(A1.v, B0.v, a10);
      if (act1) {
        size_t wb1 = (wrow + n0 + 16 + m) * 32 + hi * 8;
        B1.q[0] = *(const v4i*)&Wt[wb1];
        B1.q[1] = *(const v4i*)&Wt[wb1 + 16];
        a01 = bfwmma(A0.v, B1.v, a01);
        a11 = bfwmma(A1.v, B1.v, a11);
      }
    }
    buf ^= 1;
  }

  if (!act0) return;
  auto epi = [&](v8f& a, int nb, int mb) {
    int c = nb + m;
    if (c >= cout) return;
    float sc = 1.f, sh = 0.f, bs = 0.f;
    if (gp)   { float inv = rsqrtf(vp[c] + EPSBN); sc = gp[c] * inv; sh = bp[c] - mp[c] * sc; }
    if (bias) { bs = bias[c]; }
#pragma unroll
    for (int r = 0; r < 8; ++r) {
      int row = row0 + mb + r + 8 * hi;
      if (row >= n_out) continue;
      float x = a[r] + bs;
      if (gp)    x = sc * x + sh;
      if (resid) x += resid[(size_t)row * rstride + c];
      if (relu)  x = fmaxf(x, 0.f);
      size_t o = (size_t)row * out_stride + col_off + c;
      if (outF) outF[o] = x;
      if (outB) outB[o] = (__bf16)x;
    }
  };
  epi(a00, n0, 0);
  epi(a10, n0, 16);
  epi(a01, n0 + 16, 0);
  epi(a11, n0 + 16, 16);
}

// ---------------------------------------------------------------------------
// Weight repack: f32 [K,cin,cout] -> bf16 WMMA-B layout, zero padded.
// ---------------------------------------------------------------------------
__global__ void pack_w_k(const float* __restrict__ W, __bf16* __restrict__ Wt,
                         int K, int cin, int cout, int cinp, int coutp) {
  int nch = cinp >> 5;
  size_t n = (size_t)K * cinp * coutp;
  size_t t = (size_t)blockIdx.x * blockDim.x + threadIdx.x;
  if (t >= n) return;
  int j = (int)(t & 31);
  size_t u = t >> 5;
  int nn = (int)(u % coutp); u /= coutp;
  int cc = (int)(u % nch);
  int k  = (int)(u / nch);
  int cch = cc * 32 + j;
  float v = 0.f;
  if (cch < cin && nn < cout) v = W[((size_t)k * cin + cch) * cout + nn];
  Wt[t] = (__bf16)v;
}

// ---------------------------------------------------------------------------
// Point/voxel helpers
// ---------------------------------------------------------------------------
__global__ void fill_f32_k(float* p, float v, size_t n) {
  size_t t = (size_t)blockIdx.x * blockDim.x + threadIdx.x;
  if (t < n) p[t] = v;
}
__global__ void iota_k(int* p, int n) {
  int t = blockIdx.x * blockDim.x + threadIdx.x;
  if (t < n) p[t] = t;
}
__global__ void knbr_k(const int* __restrict__ par, const int* __restrict__ off,
                       int* __restrict__ knbr, int n) {
  int i = blockIdx.x * blockDim.x + threadIdx.x;
  if (i >= n) return;
  int o = off[i], pa = par[i];
#pragma unroll
  for (int k = 0; k < 8; ++k) knbr[(size_t)k * n + i] = (k == o) ? pa : -1;
}
__global__ void concat_k(const __bf16* __restrict__ src, int cs,
                         __bf16* __restrict__ dst, int ds_, int coloff, int n) {
  size_t t = (size_t)blockIdx.x * blockDim.x + threadIdx.x;
  if (t >= (size_t)n * cs) return;
  size_t r = t / cs; int c = (int)(t % cs);
  dst[r * ds_ + coloff + c] = src[t];
}
__global__ void p2v_scatter_k(const float* __restrict__ pf, int C,
                              const int* __restrict__ map,
                              float* __restrict__ acc, float* __restrict__ cnt, int n) {
  size_t t = (size_t)blockIdx.x * blockDim.x + threadIdx.x;
  if (t >= (size_t)n * C) return;
  size_t pidx = t / C; int c = (int)(t % C);
  int v = map[pidx];
  atomicAdd(&acc[(size_t)v * C + c], pf[t]);
  if (cnt && c == 0) atomicAdd(&cnt[v], 1.f);
}
__global__ void p2v_norm_k(const float* __restrict__ acc, const float* __restrict__ cnt,
                           float* __restrict__ outF, __bf16* __restrict__ outB,
                           int C, int Cp, int nvox) {
  size_t t = (size_t)blockIdx.x * blockDim.x + threadIdx.x;
  if (t >= (size_t)nvox * Cp) return;
  size_t v = t / Cp; int c = (int)(t % Cp);
  float x = 0.f;
  if (c < C) x = acc[v * C + c] / fmaxf(cnt[v], 1.f);
  if (outB) outB[v * Cp + c] = (__bf16)x;
  if (outF && c < C) outF[v * C + c] = x;
}
__global__ void devox_add_k(const float* __restrict__ vox, int vstride,
                            const int* __restrict__ dvi, const float* __restrict__ dvw,
                            const float* __restrict__ addf,
                            float* __restrict__ outF, __bf16* __restrict__ outB,
                            int C, int n) {
  size_t t = (size_t)blockIdx.x * blockDim.x + threadIdx.x;
  if (t >= (size_t)n * C) return;
  size_t pidx = t / C; int c = (int)(t % C);
  float s = 0.f;
#pragma unroll
  for (int k = 0; k < 8; ++k) {
    int id = dvi[pidx * 8 + k];
    s += dvw[pidx * 8 + k] * vox[(size_t)id * vstride + c];
  }
  if (addf) s += addf[t];
  if (outF) outF[t] = s;
  if (outB) outB[t] = (__bf16)s;
}

// ---------------------------------------------------------------------------
// Host orchestration
// ---------------------------------------------------------------------------
struct ConvP { const float *W, *g, *b, *m, *v; int K, cin, cout; };
struct ResP  { ConvP c1, c2, ds; bool has_ds; };
struct LinP  { const float *W, *bias, *g, *b, *m, *v; int cin, cout; bool bn; };
struct Pk    { __bf16* w; int K, cinp, coutp, nch, cout; };
struct RPk   { Pk c1, c2, ds; };

static inline int rup(int x, int a) { return (x + a - 1) / a * a; }
static inline unsigned gdiv(size_t n, int b) { return (unsigned)((n + b - 1) / b); }

extern "C" void kernel_launch(void* const* d_in, const int* in_sizes, int n_in,
                              void* d_out, int out_size, void* d_ws, size_t ws_size,
                              hipStream_t stream) {
  (void)n_in; (void)out_size; (void)ws_size;
  int ip = 0;
  const float* feats = (const float*)d_in[ip]; int npts = in_sizes[ip] / 4; ip++;

  auto CONV = [&](int K, int cin, int cout) {
    ConvP c{(const float*)d_in[ip], (const float*)d_in[ip+1], (const float*)d_in[ip+2],
            (const float*)d_in[ip+3], (const float*)d_in[ip+4], K, cin, cout};
    ip += 5; return c;
  };
  auto RES = [&](int cin, int cout) {
    ResP r; r.c1 = CONV(27, cin, cout); r.c2 = CONV(27, cout, cout);
    r.has_ds = (cin != cout); if (r.has_ds) r.ds = CONV(1, cin, cout);
    return r;
  };
  auto LIN = [&](int cin, int cout, bool bn) {
    LinP l; l.W = (const float*)d_in[ip]; l.bias = (const float*)d_in[ip+1]; ip += 2;
    if (bn) { l.g=(const float*)d_in[ip]; l.b=(const float*)d_in[ip+1];
              l.m=(const float*)d_in[ip+2]; l.v=(const float*)d_in[ip+3]; ip += 4; }
    else { l.g=l.b=l.m=l.v=nullptr; }
    l.cin=cin; l.cout=cout; l.bn=bn; return l;
  };

  ConvP stem1=CONV(27,4,32), stem2=CONV(27,32,32);
  ConvP st1d=CONV(8,32,32);    ResP st1r1=RES(32,32),   st1r2=RES(32,32);
  ConvP st2d=CONV(8,32,32);    ResP st2r1=RES(32,64),   st2r2=RES(64,64);
  ConvP st3d=CONV(8,64,64);    ResP st3r1=RES(64,128),  st3r2=RES(128,128);
  ConvP st4d=CONV(8,128,128);  ResP st4r1=RES(128,256), st4r2=RES(256,256);
  ConvP up1t=CONV(8,256,256);  ResP up1r1=RES(384,256), up1r2=RES(256,256);
  ConvP up2t=CONV(8,256,128);  ResP up2r1=RES(192,128), up2r2=RES(128,128);
  ConvP up3t=CONV(8,128,96);   ResP up3r1=RES(128,96),  up3r2=RES(96,96);
  ConvP up4t=CONV(8,96,96);    ResP up4r1=RES(128,96),  up4r2=RES(96,96);
  LinP pt0=LIN(32,256,true), pt1=LIN(256,128,true), pt2=LIN(128,96,true), cls=LIN(96,20,false);

  // maps
  const int* nbr1=(const int*)d_in[ip]; int n1=in_sizes[ip]/27; ip++;
  const int* nbr2=(const int*)d_in[ip]; int n2=in_sizes[ip]/27; ip++;
  const int* nbr4=(const int*)d_in[ip]; int n4=in_sizes[ip]/27; ip++;
  const int* nbr8=(const int*)d_in[ip]; int n8=in_sizes[ip]/27; ip++;
  const int* nbr16=(const int*)d_in[ip]; int n16=in_sizes[ip]/27; ip++;
  const int* nd2=(const int*)d_in[ip++];  const int* nd4=(const int*)d_in[ip++];
  const int* nd8=(const int*)d_in[ip++];  const int* nd16=(const int*)d_in[ip++];
  const int* par1=(const int*)d_in[ip++]; const int* off1=(const int*)d_in[ip++];
  const int* par2=(const int*)d_in[ip++]; const int* off2=(const int*)d_in[ip++];
  const int* par4=(const int*)d_in[ip++]; const int* off4=(const int*)d_in[ip++];
  const int* par8=(const int*)d_in[ip++]; const int* off8=(const int*)d_in[ip++];
  const int* p2v1=(const int*)d_in[ip++]; const int* dvi1=(const int*)d_in[ip++];
  const float* dvw1=(const float*)d_in[ip++];
  const int* p2v4=(const int*)d_in[ip++]; const int* dvi4=(const int*)d_in[ip++];
  const float* dvw4=(const float*)d_in[ip++];
  const int* p2v16=(const int*)d_in[ip++]; const int* dvi16=(const int*)d_in[ip++];
  const float* dvw16=(const float*)d_in[ip++];

  // ---- workspace bump allocator ----
  char* wsp = (char*)d_ws; size_t wo = 0;
  auto alloc = [&](size_t bytes) -> void* {
    void* r = wsp + wo; wo += (bytes + 255) & ~(size_t)255; return r;
  };

  // ---- pack all weights to bf16 WMMA-B layout ----
  auto packW = [&](const float* W, int K, int cin, int cout) {
    Pk pk; pk.K=K; pk.cinp=rup(cin,32); pk.coutp=rup(cout,16); pk.nch=pk.cinp/32; pk.cout=cout;
    size_t n = (size_t)K * pk.cinp * pk.coutp;
    pk.w = (__bf16*)alloc(n * 2);
    pack_w_k<<<gdiv(n,256),256,0,stream>>>(W, pk.w, K, cin, cout, pk.cinp, pk.coutp);
    return pk;
  };
  auto packC = [&](const ConvP& c) { return packW(c.W, c.K, c.cin, c.cout); };
  auto packR = [&](const ResP& r)  {
    RPk k; k.c1 = packC(r.c1); k.c2 = packC(r.c2);
    if (r.has_ds) k.ds = packC(r.ds); return k;
  };
  Pk  pkStem1=packC(stem1), pkStem2=packC(stem2);
  Pk  pkS1d=packC(st1d), pkS2d=packC(st2d), pkS3d=packC(st3d), pkS4d=packC(st4d);
  RPk pk1r1=packR(st1r1), pk1r2=packR(st1r2), pk2r1=packR(st2r1), pk2r2=packR(st2r2);
  RPk pk3r1=packR(st3r1), pk3r2=packR(st3r2), pk4r1=packR(st4r1), pk4r2=packR(st4r2);
  Pk  pkU1=packC(up1t), pkU2=packC(up2t), pkU3=packC(up3t), pkU4=packC(up4t);
  RPk pkU1r1=packR(up1r1), pkU1r2=packR(up1r2), pkU2r1=packR(up2r1), pkU2r2=packR(up2r2);
  RPk pkU3r1=packR(up3r1), pkU3r2=packR(up3r2), pkU4r1=packR(up4r1), pkU4r2=packR(up4r2);
  Pk  pkPt0=packW(pt0.W,1,32,256), pkPt1=packW(pt1.W,1,256,128);
  Pk  pkPt2=packW(pt2.W,1,128,96), pkCls=packW(cls.W,1,96,20);

  // ---- buffers ----
  size_t maxI = (size_t)(npts > n1 ? npts : n1);
  int* iota = (int*)alloc(maxI * 4);
  iota_k<<<gdiv(maxI,256),256,0,stream>>>(iota, (int)maxI);
  int*   knbr  = (int*)alloc((size_t)n1 * 8 * 4);
  float* cnt1  = (float*)alloc((size_t)n1 * 4);
  float* cnt4  = (float*)alloc((size_t)n4 * 4);
  float* cnt16 = (float*)alloc((size_t)n16 * 4);
  float* accB  = (float*)alloc((size_t)n1 * 256 * 4);
  float* tmpS  = (float*)alloc(maxI * 256 * 4);
  __bf16* tmpH = (__bf16*)alloc((size_t)n1 * 256 * 2);
  float* scrAf = (float*)alloc((size_t)n1 * 256 * 4);
  __bf16* scrAb= (__bf16*)alloc((size_t)n1 * 256 * 2);
  float* scrBf = (float*)alloc((size_t)n1 * 256 * 4);
  __bf16* scrBb= (__bf16*)alloc((size_t)n1 * 256 * 2);
  __bf16* x0in = (__bf16*)alloc((size_t)n1 * 32 * 2);
  __bf16* x1in = (__bf16*)alloc((size_t)n1 * 32 * 2);
  float* x0f=(float*)alloc((size_t)n1*32*4);   __bf16* x0b=(__bf16*)alloc((size_t)n1*32*2);
  float* x1f=(float*)alloc((size_t)n2*32*4);   __bf16* x1b=(__bf16*)alloc((size_t)n2*32*2);
  float* x2f=(float*)alloc((size_t)n4*64*4);   __bf16* x2b=(__bf16*)alloc((size_t)n4*64*2);
  float* x3f=(float*)alloc((size_t)n8*128*4);  __bf16* x3b=(__bf16*)alloc((size_t)n8*128*2);
  float* x4f=(float*)alloc((size_t)n16*256*4); __bf16* x4b=(__bf16*)alloc((size_t)n16*256*2);
  float* z0f=(float*)alloc((size_t)npts*32*4); __bf16* z0b=(__bf16*)alloc((size_t)npts*32*2);
  float* z1f=(float*)alloc((size_t)npts*256*4);__bf16* z1b=(__bf16*)alloc((size_t)npts*256*2);
  float* z2f=(float*)alloc((size_t)npts*128*4);__bf16* z2b=(__bf16*)alloc((size_t)npts*128*2);
  __bf16* z3b=(__bf16*)alloc((size_t)npts*96*2);
  __bf16* v16b=(__bf16*)alloc((size_t)n16*256*2);
  __bf16* v4b =(__bf16*)alloc((size_t)n4*128*2);
  __bf16* wide1=(__bf16*)alloc((size_t)n8*384*2);
  __bf16* wide2=(__bf16*)alloc((size_t)n4*192*2);
  __bf16* wide3=(__bf16*)alloc((size_t)n2*128*2);
  __bf16* wide4=(__bf16*)alloc((size_t)n1*128*2);
  float* y1f=(float*)alloc((size_t)n8*256*4);  __bf16* y1b=(__bf16*)alloc((size_t)n8*256*2);
  float* y2f=(float*)alloc((size_t)n4*128*4);  __bf16* y2b=(__bf16*)alloc((size_t)n4*128*2);
  float* y3f=(float*)alloc((size_t)n2*96*4);   __bf16* y3b=(__bf16*)alloc((size_t)n2*96*2);
  float* y4f=(float*)alloc((size_t)n1*96*4);   __bf16* y4b=(__bf16*)alloc((size_t)n1*96*2);

  // ---- small launch helpers ----
  auto fill = [&](float* pdst, size_t n) {
    fill_f32_k<<<gdiv(n,256),256,0,stream>>>(pdst, 0.f, n);
  };
  auto conv = [&](const __bf16* fb, int instr, const Pk& pk, const int* nb, int pitch, int n,
                  float* oF, __bf16* oB, int ostr, int coff,
                  const float* bias, const float* g, const float* b,
                  const float* m, const float* v,
                  const float* rs, int rstr, int relu) {
    dim3 gr((unsigned)((n + 31) / 32), (unsigned)((pk.coutp + 127) / 128));
    spconv_wmma<<<gr,128,0,stream>>>(fb, instr, pk.w, nb, pitch, pk.K, pk.nch, n,
                                     pk.cout, pk.coutp, oF, oB, ostr, coff,
                                     bias, g, b, m, v, rs, rstr, relu);
  };
  auto cbr = [&](const __bf16* fb, int instr, const ConvP& cp, const Pk& pk,
                 const int* nb, int pitch, int n, float* oF, __bf16* oB) {
    conv(fb, instr, pk, nb, pitch, n, oF, oB, cp.cout, 0,
         nullptr, cp.g, cp.b, cp.m, cp.v, nullptr, 0, 1);
  };
  auto res = [&](const __bf16* fb, const float* ff, int cin, const int* nb, int pitch, int n,
                 const ResP& rp, const RPk& rk, float* oF, __bf16* oB) {
    int cout = rp.c1.cout;
    conv(fb, cin, rk.c1, nb, pitch, n, nullptr, tmpH, cout, 0,
         nullptr, rp.c1.g, rp.c1.b, rp.c1.m, rp.c1.v, nullptr, 0, 1);
    const float* rs;
    if (rp.has_ds) {
      conv(fb, cin, rk.ds, iota, n, n, tmpS, nullptr, cout, 0,
           nullptr, rp.ds.g, rp.ds.b, rp.ds.m, rp.ds.v, nullptr, 0, 0);
      rs = tmpS;
    } else rs = ff;
    conv(tmpH, cout, rk.c2, nb, pitch, n, oF, oB, cout, 0,
         nullptr, rp.c2.g, rp.c2.b, rp.c2.m, rp.c2.v, rs, cout, 1);
  };
  auto lin = [&](const __bf16* fb, int instr, const LinP& lp, const Pk& pk,
                 float* oF, __bf16* oB, int n) {
    conv(fb, instr, pk, iota, n, n, oF, oB, lp.cout, 0,
         lp.bias, lp.g, lp.b, lp.m, lp.v, nullptr, 0, lp.bn ? 1 : 0);
  };
  auto scat = [&](const float* pf, int C, const int* map, float* a, float* c, int n) {
    p2v_scatter_k<<<gdiv((size_t)n*C,256),256,0,stream>>>(pf, C, map, a, c, n);
  };
  auto norm = [&](const float* a, const float* c, float* oF, __bf16* oB, int C, int Cp, int nv) {
    p2v_norm_k<<<gdiv((size_t)nv*Cp,256),256,0,stream>>>(a, c, oF, oB, C, Cp, nv);
  };
  auto devox = [&](const float* vox, int C, const int* di, const float* dw,
                   const float* add, float* oF, __bf16* oB) {
    devox_add_k<<<gdiv((size_t)npts*C,256),256,0,stream>>>(vox, C, di, dw, add, oF, oB, C, npts);
  };
  auto deconv = [&](const __bf16* coarse, int cin, const ConvP& cp, const Pk& pk,
                    const int* par, const int* off, int nf, __bf16* wide, int ctot) {
    knbr_k<<<gdiv((size_t)nf,256),256,0,stream>>>(par, off, knbr, nf);
    conv(coarse, cin, pk, knbr, nf, nf, nullptr, wide, ctot, 0,
         nullptr, cp.g, cp.b, cp.m, cp.v, nullptr, 0, 1);
  };
  auto concat = [&](const __bf16* src, int cs, __bf16* dst, int ds_, int coff, int n) {
    concat_k<<<gdiv((size_t)n*cs,256),256,0,stream>>>(src, cs, dst, ds_, coff, n);
  };

  // ======================= forward graph =======================
  // x0 = p2v(feats); stems
  fill(accB, (size_t)n1*4); fill(cnt1, (size_t)n1);
  scat(feats, 4, p2v1, accB, cnt1, npts);
  norm(accB, cnt1, nullptr, x0in, 4, 32, n1);
  cbr(x0in, 32, stem1, pkStem1, nbr1, n1, n1, nullptr, tmpH);
  cbr(tmpH, 32, stem2, pkStem2, nbr1, n1, n1, x0f, x0b);
  // z0 = dv(x0)
  devox(x0f, 32, dvi1, dvw1, nullptr, z0f, z0b);
  // x1 voxels = p2v(z0)
  fill(accB, (size_t)n1*32);
  scat(z0f, 32, p2v1, accB, nullptr, npts);
  norm(accB, cnt1, nullptr, x1in, 32, 32, n1);
  // stage 1
  cbr(x1in, 32, st1d, pkS1d, nd2, n2, n2, scrAf, scrAb);
  res(scrAb, scrAf, 32, nbr2, n2, n2, st1r1, pk1r1, scrBf, scrBb);
  res(scrBb, scrBf, 32, nbr2, n2, n2, st1r2, pk1r2, x1f, x1b);
  // stage 2
  cbr(x1b, 32, st2d, pkS2d, nd4, n4, n4, scrAf, scrAb);
  res(scrAb, scrAf, 32, nbr4, n4, n4, st2r1, pk2r1, scrBf, scrBb);
  res(scrBb, scrBf, 64, nbr4, n4, n4, st2r2, pk2r2, x2f, x2b);
  // stage 3
  cbr(x2b, 64, st3d, pkS3d, nd8, n8, n8, scrAf, scrAb);
  res(scrAb, scrAf, 64, nbr8, n8, n8, st3r1, pk3r1, scrBf, scrBb);
  res(scrBb, scrBf, 128, nbr8, n8, n8, st3r2, pk3r2, x3f, x3b);
  // stage 4
  cbr(x3b, 128, st4d, pkS4d, nd16, n16, n16, scrAf, scrAb);
  res(scrAb, scrAf, 128, nbr16, n16, n16, st4r1, pk4r1, scrBf, scrBb);
  res(scrBb, scrBf, 256, nbr16, n16, n16, st4r2, pk4r2, x4f, x4b);
  // z1 = dv(x4) + lbr(z0, pt0)
  lin(z0b, 32, pt0, pkPt0, tmpS, nullptr, npts);
  devox(x4f, 256, dvi16, dvw16, tmpS, z1f, z1b);
  // v16 = p2v(z1)
  fill(accB, (size_t)n16*256); fill(cnt16, (size_t)n16);
  scat(z1f, 256, p2v16, accB, cnt16, npts);
  norm(accB, cnt16, nullptr, v16b, 256, 256, n16);
  // up1
  deconv(v16b, 256, up1t, pkU1, par8, off8, n8, wide1, 384);
  concat(x3b, 128, wide1, 384, 256, n8);
  res(wide1, nullptr, 384, nbr8, n8, n8, up1r1, pkU1r1, scrAf, scrAb);
  res(scrAb, scrAf, 256, nbr8, n8, n8, up1r2, pkU1r2, y1f, y1b);
  // up2
  deconv(y1b, 256, up2t, pkU2, par4, off4, n4, wide2, 192);
  concat(x2b, 64, wide2, 192, 128, n4);
  res(wide2, nullptr, 192, nbr4, n4, n4, up2r1, pkU2r1, scrAf, scrAb);
  res(scrAb, scrAf, 128, nbr4, n4, n4, up2r2, pkU2r2, y2f, y2b);
  // z2 = dv(y2) + lbr(z1, pt1)
  lin(z1b, 256, pt1, pkPt1, tmpS, nullptr, npts);
  devox(y2f, 128, dvi4, dvw4, tmpS, z2f, z2b);
  // v4 = p2v(z2)
  fill(accB, (size_t)n4*128); fill(cnt4, (size_t)n4);
  scat(z2f, 128, p2v4, accB, cnt4, npts);
  norm(accB, cnt4, nullptr, v4b, 128, 128, n4);
  // up3
  deconv(v4b, 128, up3t, pkU3, par2, off2, n2, wide3, 128);
  concat(x1b, 32, wide3, 128, 96, n2);
  res(wide3, nullptr, 128, nbr2, n2, n2, up3r1, pkU3r1, scrAf, scrAb);
  res(scrAb, scrAf, 96, nbr2, n2, n2, up3r2, pkU3r2, y3f, y3b);
  // up4
  deconv(y3b, 96, up4t, pkU4, par1, off1, n1, wide4, 128);
  concat(x0b, 32, wide4, 128, 96, n1);
  res(wide4, nullptr, 128, nbr1, n1, n1, up4r1, pkU4r1, scrAf, scrAb);
  res(scrAb, scrAf, 96, nbr1, n1, n1, up4r2, pkU4r2, y4f, y4b);
  // z3 = dv(y4) + lbr(z2, pt2)
  lin(z2b, 128, pt2, pkPt2, tmpS, nullptr, npts);
  devox(y4f, 96, dvi1, dvw1, tmpS, nullptr, z3b);
  // logits = z3 @ cls.W + cls.bias
  lin(z3b, 96, cls, pkCls, (float*)d_out, nullptr, npts);
}
